// GraphAutoencoder_27212912787816
// MI455X (gfx1250) — compile-verified, matrix operand order checked by reading, not run on hardware
//
#include <hip/hip_runtime.h>
#include <hip/hip_bf16.h>

#define N_NODES 50000
#define N_EDGES 1000000
#define NUM_IDS 256
#define EMB_D   8
#define HID     32
#define LAT     32
#define HEADS   4
#define IN_CH   16
#define GAT_IN  23          // EMB + (IN_CH-1)
#define BN_EPS  1e-5f

typedef __attribute__((ext_vector_type(16))) _Float16 v16h;
typedef __attribute__((ext_vector_type(8)))  _Float16 v8h;
typedef __attribute__((ext_vector_type(8)))  float    v8f;

// ---------------------------------------------------------------------------
// CDNA5 16-bit 16x32 fragment layout (ISA 7.12.2): lane half selects K-half;
// a lane's 16 elements are two contiguous 8-element (16-byte) runs:
//   k_local in [8*half, 8*half+8)  and  [16+8*half, 16+8*half+8)
// With A row-major and B stored transposed ([col][k], zero-padded), every
// operand fragment is exactly two global_load_b128 per lane.
// ---------------------------------------------------------------------------
__device__ __forceinline__ v16h frag_from2(v8h lo, v8h hi) {
    return __builtin_shufflevector(lo, hi, 0, 1, 2, 3, 4, 5, 6, 7,
                                           8, 9, 10, 11, 12, 13, 14, 15);
}

__device__ __forceinline__ void atomicMaxF(float* addr, float val) {
    int old = __float_as_int(*addr);
    while (__int_as_float(old) < val) {
        int assumed = old;
        old = atomicCAS((int*)addr, assumed, __float_as_int(val));
        if (old == assumed) break;
    }
}

// Pre-swizzle weights: W f32 [Kact, Ncols] -> Bt f16 [Npad, Kpad], zero-padded.
__global__ void weight_prep_kernel(const float* __restrict__ W, int Kact, int Ncols,
                                   int Kpad, int Npad, _Float16* __restrict__ Bt) {
    int i = blockIdx.x * blockDim.x + threadIdx.x;
    if (i >= Npad * Kpad) return;
    int c = i / Kpad, k = i % Kpad;
    float v = (c < Ncols && k < Kact) ? W[(size_t)k * Ncols + c] : 0.f;
    Bt[i] = (_Float16)v;
}

// ---------------------------------------------------------------------------
// Wave-per-16x16-tile GEMM: C[M,Ncols] = A16[M,Kpad] @ Bt^T, Bt f16 [Npad,Kpad]
// Inner loop: 4x global_load_b128 + 1x v_wmma_f32_16x16x32_f16.
// ---------------------------------------------------------------------------
__global__ void gemm_bt_kernel(const _Float16* __restrict__ A, int M, int lda,
                               int Kpad, const _Float16* __restrict__ Bt, int Ncols,
                               const float* __restrict__ bias,
                               float* __restrict__ Cf, _Float16* __restrict__ Ch,
                               int ldc) {
    const int lane = threadIdx.x & 31;
    const int half = lane >> 4;
    const int l15 = lane & 15;
    const int mt = blockIdx.x * 16;
    const int nt = blockIdx.y * 16;
    int ar = mt + l15; if (ar > M - 1) ar = M - 1;   // clamp: loads valid, stores guarded
    const _Float16* Arow = A + (size_t)ar * lda + 8 * half;
    const _Float16* Brow = Bt + (size_t)(nt + l15) * Kpad + 8 * half;
    v8f acc = {};
    for (int k0 = 0; k0 < Kpad; k0 += 32) {
        v8h alo = *(const v8h*)(Arow + k0);
        v8h ahi = *(const v8h*)(Arow + k0 + 16);
        v8h blo = *(const v8h*)(Brow + k0);
        v8h bhi = *(const v8h*)(Brow + k0 + 16);
        acc = __builtin_amdgcn_wmma_f32_16x16x32_f16(false, frag_from2(alo, ahi),
                                                     false, frag_from2(blo, bhi),
                                                     (short)0, acc, false, false);
    }
    const int col = nt + l15;
    if (col >= Ncols) return;
    float bv = bias ? bias[col] : 0.f;
#pragma unroll
    for (int r = 0; r < 8; r++) {
        int row = mt + r + 8 * half;                 // C/D layout: VGPR r -> M=r / M=8+r
        if (row < M) {
            float v = acc[r] + bv;
            if (Cf) Cf[(size_t)row * ldc + col] = v;
            if (Ch) Ch[(size_t)row * ldc + col] = (_Float16)v;
        }
    }
}

// ---------------------------------------------------------------------------
// Edge MLP layer 1: t1 = ef @ W1 + b1, ef[e] = [zs, zd, zs*zd, |zs-zd|].
// ef is synthesized in registers from eight float4 loads of the two z rows
// (never materialized: saves ~1GB of HBM round-trip at 23.3 TB/s).
// ---------------------------------------------------------------------------
__device__ __forceinline__ v16h pack4(float4 p0, float4 p1, float4 p2, float4 p3) {
    v16h a;
    a[0]=(_Float16)p0.x;  a[1]=(_Float16)p0.y;  a[2]=(_Float16)p0.z;  a[3]=(_Float16)p0.w;
    a[4]=(_Float16)p1.x;  a[5]=(_Float16)p1.y;  a[6]=(_Float16)p1.z;  a[7]=(_Float16)p1.w;
    a[8]=(_Float16)p2.x;  a[9]=(_Float16)p2.y;  a[10]=(_Float16)p2.z; a[11]=(_Float16)p2.w;
    a[12]=(_Float16)p3.x; a[13]=(_Float16)p3.y; a[14]=(_Float16)p3.z; a[15]=(_Float16)p3.w;
    return a;
}
__device__ __forceinline__ float4 mul4(float4 a, float4 b) {
    return make_float4(a.x*b.x, a.y*b.y, a.z*b.z, a.w*b.w);
}
__device__ __forceinline__ float4 ad4(float4 a, float4 b) {   // |a-b| w/o fabs (ISel bug)
    return make_float4(fmaxf(a.x-b.x, b.x-a.x), fmaxf(a.y-b.y, b.y-a.y),
                       fmaxf(a.z-b.z, b.z-a.z), fmaxf(a.w-b.w, b.w-a.w));
}

__global__ void edge_gemm1_kernel(const float* __restrict__ z,
                                  const int* __restrict__ esrc,
                                  const int* __restrict__ edst,
                                  const _Float16* __restrict__ Bt,  // ed1 W^T f16 [128,128]
                                  const float* __restrict__ bias,
                                  _Float16* __restrict__ out) {
    const int lane = threadIdx.x & 31;
    const int half = lane >> 4;
    const int l15 = lane & 15;
    const int mt = blockIdx.x * 16;
    const int nt = blockIdx.y * 16;
    int e = mt + l15; if (e > N_EDGES - 1) e = N_EDGES - 1;
    const float* zs = z + (size_t)esrc[e] * LAT;
    const float* zd = z + (size_t)edst[e] * LAT;
    const int o1 = 8 * half, o2 = 16 + 8 * half;
    float4 s0 = *(const float4*)(zs + o1);
    float4 s1 = *(const float4*)(zs + o1 + 4);
    float4 s2 = *(const float4*)(zs + o2);
    float4 s3 = *(const float4*)(zs + o2 + 4);
    float4 d0 = *(const float4*)(zd + o1);
    float4 d1 = *(const float4*)(zd + o1 + 4);
    float4 d2 = *(const float4*)(zd + o2);
    float4 d3 = *(const float4*)(zd + o2 + 4);
    v16h aS = pack4(s0, s1, s2, s3);                              // k in [0,32)
    v16h aD = pack4(d0, d1, d2, d3);                              // k in [32,64)
    v16h aP = pack4(mul4(s0,d0), mul4(s1,d1), mul4(s2,d2), mul4(s3,d3));   // [64,96)
    v16h aA = pack4(ad4(s0,d0), ad4(s1,d1), ad4(s2,d2), ad4(s3,d3));       // [96,128)

    const _Float16* Brow = Bt + (size_t)(nt + l15) * 128 + 8 * half;
    v8f acc = {};
    v16h b;
    b = frag_from2(*(const v8h*)(Brow + 0),  *(const v8h*)(Brow + 16));
    acc = __builtin_amdgcn_wmma_f32_16x16x32_f16(false, aS, false, b, (short)0, acc, false, false);
    b = frag_from2(*(const v8h*)(Brow + 32), *(const v8h*)(Brow + 48));
    acc = __builtin_amdgcn_wmma_f32_16x16x32_f16(false, aD, false, b, (short)0, acc, false, false);
    b = frag_from2(*(const v8h*)(Brow + 64), *(const v8h*)(Brow + 80));
    acc = __builtin_amdgcn_wmma_f32_16x16x32_f16(false, aP, false, b, (short)0, acc, false, false);
    b = frag_from2(*(const v8h*)(Brow + 96), *(const v8h*)(Brow + 112));
    acc = __builtin_amdgcn_wmma_f32_16x16x32_f16(false, aA, false, b, (short)0, acc, false, false);

    const int col = nt + l15;
    float bv = bias[col];
#pragma unroll
    for (int r = 0; r < 8; r++) {
        int row = mt + r + 8 * half;
        if (row < N_EDGES) out[(size_t)row * 128 + col] = (_Float16)(acc[r] + bv);
    }
}

// ---------------------------------------------------------------------------
// Elementwise / reduction support kernels
// ---------------------------------------------------------------------------
__global__ void fill_kernel(float* p, long long n, float v) {
    long long i = (long long)blockIdx.x * blockDim.x + threadIdx.x;
    if (i < n) p[i] = v;
}

__global__ void embed_concat_kernel(const int* __restrict__ ids,
                                    const float* __restrict__ feats,
                                    const float* __restrict__ emb,
                                    _Float16* __restrict__ x16) {
    long long idx = (long long)blockIdx.x * blockDim.x + threadIdx.x;
    if (idx >= (long long)N_NODES * 32) return;
    int n = (int)(idx >> 5), c = (int)(idx & 31);
    float v = 0.f;
    if (c < EMB_D)       v = emb[(size_t)ids[n] * EMB_D + c];
    else if (c < GAT_IN) v = feats[(size_t)n * (IN_CH - 1) + (c - EMB_D)];
    x16[idx] = (_Float16)v;   // zero-padded 23 -> 32 for WMMA K tiling
}

// BN training-mode stats: one block per channel, strided row loop + LDS reduce
__global__ void bn_stats_kernel(const float* __restrict__ Xf,
                                const _Float16* __restrict__ Xh,
                                long long rows, int ld, float* mean, float* rstd) {
    int c = blockIdx.x;
    __shared__ float ss[256], sq[256];
    float s = 0.f, q = 0.f;
    for (long long r = threadIdx.x; r < rows; r += blockDim.x) {
        float v = Xf ? Xf[r * ld + c] : (float)Xh[r * ld + c];
        s += v; q += v * v;
    }
    ss[threadIdx.x] = s; sq[threadIdx.x] = q;
    __syncthreads();
    for (int o = 128; o > 0; o >>= 1) {
        if ((int)threadIdx.x < o) { ss[threadIdx.x] += ss[threadIdx.x + o];
                                    sq[threadIdx.x] += sq[threadIdx.x + o]; }
        __syncthreads();
    }
    if (threadIdx.x == 0) {
        float mu = ss[0] / (float)rows;
        float var = sq[0] / (float)rows - mu * mu;
        mean[c] = mu;
        rstd[c] = rsqrtf(var + BN_EPS);
    }
}

__global__ void bn_apply_kernel(const float* __restrict__ Xf,
                                const _Float16* __restrict__ Xh,
                                long long total, int C,
                                const float* __restrict__ mean,
                                const float* __restrict__ rstd,
                                const float* __restrict__ g,
                                const float* __restrict__ bb,
                                int relu, float* Of, _Float16* Oh) {
    long long i = (long long)blockIdx.x * blockDim.x + threadIdx.x;
    if (i >= total) return;
    int c = (int)(i % C);
    float v = Xf ? Xf[i] : (float)Xh[i];
    v = (v - mean[c]) * rstd[c] * g[c] + bb[c];
    if (relu && v < 0.f) v = 0.f;
    if (Of) Of[i] = v;
    if (Oh) Oh[i] = (_Float16)v;
}

// ---------------------------------------------------------------------------
// GAT message passing (segment softmax via 2 atomic passes)
// ---------------------------------------------------------------------------
__global__ void att_coef_kernel(const float* __restrict__ h, int H, int C,
                                const float* __restrict__ a_src,
                                const float* __restrict__ a_dst,
                                float* as, float* ad) {
    long long i = (long long)blockIdx.x * blockDim.x + threadIdx.x;
    if (i >= (long long)N_NODES * H) return;
    int n = (int)(i / H), hh = (int)(i % H);
    const float* hp = h + ((size_t)n * H + hh) * C;
    float s1 = 0.f, s2 = 0.f;
    for (int c = 0; c < C; c++) { s1 += hp[c] * a_src[hh * C + c];
                                  s2 += hp[c] * a_dst[hh * C + c]; }
    as[i] = s1; ad[i] = s2;
}

__global__ void gat_edge_max_kernel(const int* __restrict__ esrc,
                                    const int* __restrict__ edst, int H,
                                    const float* __restrict__ as,
                                    const float* __restrict__ ad,
                                    float* elr, float* m) {
    long long i = (long long)blockIdx.x * blockDim.x + threadIdx.x;
    long long total = (long long)(N_EDGES + N_NODES) * H;
    if (i >= total) return;
    long long e = i / H; int hh = (int)(i % H);
    int s = (e < N_EDGES) ? esrc[e] : (int)(e - N_EDGES);   // self loops appended
    int d = (e < N_EDGES) ? edst[e] : (int)(e - N_EDGES);
    float x = as[(size_t)s * H + hh] + ad[(size_t)d * H + hh];
    x = (x > 0.f) ? x : 0.2f * x;                           // leaky_relu
    elr[i] = x;
    atomicMaxF(&m[(size_t)d * H + hh], x);
}

__global__ void gat_edge_accum_kernel(const int* __restrict__ esrc,
                                      const int* __restrict__ edst, int H, int C,
                                      const float* __restrict__ elr,
                                      const float* __restrict__ m,
                                      const float* __restrict__ hfeat,
                                      float* ssum, float* agg) {
    long long i = (long long)blockIdx.x * blockDim.x + threadIdx.x;
    long long total = (long long)(N_EDGES + N_NODES) * H;
    if (i >= total) return;
    long long e = i / H; int hh = (int)(i % H);
    int s = (e < N_EDGES) ? esrc[e] : (int)(e - N_EDGES);
    int d = (e < N_EDGES) ? edst[e] : (int)(e - N_EDGES);
    float ex = __expf(elr[i] - m[(size_t)d * H + hh]);
    atomicAdd(&ssum[(size_t)d * H + hh], ex);
    const float* hr = hfeat + ((size_t)s * H + hh) * C;
    float* ar = agg + ((size_t)d * H + hh) * C;
    for (int c = 0; c < C; c++) atomicAdd(&ar[c], hr[c] * ex);
}

__global__ void gat_finalize_kernel(const float* __restrict__ agg,
                                    const float* __restrict__ ssum,
                                    const float* __restrict__ bias,
                                    int H, int C, int do_sigmoid, float* out) {
    long long i = (long long)blockIdx.x * blockDim.x + threadIdx.x;
    long long total = (long long)N_NODES * H * C;
    if (i >= total) return;
    int HC = H * C;
    int n = (int)(i / HC), j = (int)(i % HC), hh = j / C;
    float v = agg[i] / (ssum[(size_t)n * H + hh] + 1e-16f) + bias[j];
    if (do_sigmoid) v = 1.f / (1.f + __expf(-v));
    out[i] = v;
}

__global__ void residual_kernel(const float* a, const float* b,
                                float* of, _Float16* oh) {
    long long i = (long long)blockIdx.x * blockDim.x + threadIdx.x;
    if (i >= (long long)N_NODES * HID) return;
    float v = a[i] + b[i];
    of[i] = v; oh[i] = (_Float16)v;
}

__global__ void reparam_kernel(const float* __restrict__ mu,
                               const float* __restrict__ lv,
                               const float* __restrict__ eps,
                               float* z, _Float16* z16, float* kl) {
    long long n = (long long)N_NODES * LAT;
    long long i = (long long)blockIdx.x * blockDim.x + threadIdx.x;
    float contrib = 0.f;
    if (i < n) {
        float m = mu[i], l = lv[i];
        float zz = m + eps[i] * __expf(0.5f * l);
        z[i] = zz; z16[i] = (_Float16)zz;
        contrib = -0.5f * (1.f + l - m * m - __expf(l)) / (float)n;
    }
    __shared__ float ss[256];
    ss[threadIdx.x] = contrib;
    __syncthreads();
    for (int o = 128; o > 0; o >>= 1) {
        if ((int)threadIdx.x < o) ss[threadIdx.x] += ss[threadIdx.x + o];
        __syncthreads();
    }
    if (threadIdx.x == 0) atomicAdd(kl, ss[0]);
}

__global__ void edge_out_kernel(const _Float16* __restrict__ e3,
                                const float* __restrict__ W4,
                                const float* __restrict__ b4, float* out) {
    int e = blockIdx.x * blockDim.x + threadIdx.x;
    if (e >= N_EDGES) return;
    const _Float16* r = e3 + (size_t)e * 32;
    float s = b4[0];
#pragma unroll
    for (int c = 0; c < 32; c++) s += (float)r[c] * W4[c];
    out[e] = 1.f / (1.f + __expf(-s));
}

// ---------------------------------------------------------------------------
// Host orchestration
// ---------------------------------------------------------------------------
static inline int cdiv(long long a, long long b) { return (int)((a + b - 1) / b); }

extern "C" void kernel_launch(void* const* d_in, const int* in_sizes, int n_in,
                              void* d_out, int out_size, void* d_ws, size_t ws_size,
                              hipStream_t stream) {
    (void)in_sizes; (void)n_in; (void)out_size; (void)ws_size;

    const int*   node_ids   = (const int*)d_in[0];
    const float* node_feats = (const float*)d_in[1];
    const int*   edge_src   = (const int*)d_in[2];
    const int*   edge_dst   = edge_src + N_EDGES;
    const float* eps        = (const float*)d_in[3];

    int pi = 4;
    auto P = [&]() { return (const float*)d_in[pi++]; };
    const float* emb = P();
    const float *e1W = P(), *e1as = P(), *e1ad = P(), *e1b = P();
    const float *bn1g = P(), *bn1b = P();
    const float *e2W = P(), *e2as = P(), *e2ad = P(), *e2b = P();
    const float *bn2g = P(), *bn2b = P();
    const float *e3W = P(), *e3as = P(), *e3ad = P(), *e3b = P();
    const float *bn3g = P(), *bn3b = P();
    const float *zmW = P(), *zmB = P();
    const float *zvW = P(), *zvB = P();
    const float *d1W = P(), *d1as = P(), *d1ad = P(), *d1b = P();
    const float *dbn1g = P(), *dbn1b = P();
    const float *d2W = P(), *d2as = P(), *d2ad = P(), *d2b = P();
    const float *dbn2g = P(), *dbn2b = P();
    const float *d3W = P(), *d3as = P(), *d3ad = P(), *d3b = P();
    const float *cidW = P(), *cidB = P();
    const float *ed1W = P(), *ed1b = P(), *edbn1g = P(), *edbn1b = P();
    const float *ed2W = P(), *ed2b = P(), *edbn2g = P(), *edbn2b = P();
    const float *ed3W = P(), *ed3b = P(), *edbn3g = P(), *edbn3b = P();
    const float *ed4W = P(), *ed4b = P();

    // output regions: cont [N,15] | canid [N,256] | edge_prob [E] | kl [1]
    float* out_cont  = (float*)d_out;
    float* out_canid = out_cont + (size_t)N_NODES * 15;
    float* out_edge  = out_canid + (size_t)N_NODES * 256;
    float* out_kl    = out_edge + N_EDGES;

    // bump allocator over d_ws (same layout every call -> deterministic)
    char* wp = (char*)d_ws;
    auto A = [&](size_t bytes) -> void* {
        void* p = wp; wp += (bytes + 255) & ~(size_t)255; return p;
    };
    _Float16* x16    = (_Float16*)A((size_t)N_NODES * 32 * 2);
    _Float16* actA   = (_Float16*)A((size_t)N_NODES * 128 * 2);
    _Float16* actB   = (_Float16*)A((size_t)N_NODES * 128 * 2);
    float* t_h   = (float*)A((size_t)N_NODES * 128 * 4);
    float* agg   = (float*)A((size_t)N_NODES * 128 * 4);
    float* as_b  = (float*)A((size_t)N_NODES * HEADS * 4);
    float* ad_b  = (float*)A((size_t)N_NODES * HEADS * 4);
    float* m_b   = (float*)A((size_t)N_NODES * HEADS * 4);
    float* s_b   = (float*)A((size_t)N_NODES * HEADS * 4);
    float* elr   = (float*)A((size_t)(N_EDGES + N_NODES) * HEADS * 4);
    float* bmean = (float*)A(256 * 4);
    float* brstd = (float*)A(256 * 4);
    float* x2f   = (float*)A((size_t)N_NODES * HID * 4);
    float* x3f   = (float*)A((size_t)N_NODES * HID * 4);
    float* hres  = (float*)A((size_t)N_NODES * HID * 4);
    _Float16* hres16 = (_Float16*)A((size_t)N_NODES * HID * 2);
    float* mu    = (float*)A((size_t)N_NODES * LAT * 4);
    float* lvar  = (float*)A((size_t)N_NODES * LAT * 4);
    float* zf    = (float*)A((size_t)N_NODES * LAT * 4);
    _Float16* z16 = (_Float16*)A((size_t)N_NODES * LAT * 2);
    _Float16* t1 = (_Float16*)A((size_t)N_EDGES * 128 * 2);
    _Float16* t2 = (_Float16*)A((size_t)N_EDGES * 64 * 2);
    _Float16* t3 = (_Float16*)A((size_t)N_EDGES * 32 * 2);

    // pre-swizzled fp16 weight buffers (transposed, zero-padded)
    auto prep = [&](const float* W, int Kact, int Ncols, int Kpad, int Npad) -> _Float16* {
        _Float16* Bt = (_Float16*)A((size_t)Npad * Kpad * 2);
        weight_prep_kernel<<<cdiv((long long)Npad * Kpad, 256), 256, 0, stream>>>(
            W, Kact, Ncols, Kpad, Npad, Bt);
        return Bt;
    };
    _Float16* e1t  = prep(e1W,  GAT_IN, 128, 32, 128);
    _Float16* e2t  = prep(e2W,  128, 32, 128, 32);
    _Float16* e3t  = prep(e3W,  32, 32, 32, 32);
    _Float16* zmt  = prep(zmW,  32, 32, 32, 32);
    _Float16* zvt  = prep(zvW,  32, 32, 32, 32);
    _Float16* d1t  = prep(d1W,  32, 128, 32, 128);
    _Float16* d2t  = prep(d2W,  128, 32, 128, 32);
    _Float16* d3t  = prep(d3W,  32, 15, 32, 16);
    _Float16* cidt = prep(cidW, 32, 256, 32, 256);
    _Float16* ed1t = prep(ed1W, 128, 128, 128, 128);
    _Float16* ed2t = prep(ed2W, 128, 64, 128, 64);
    _Float16* ed3t = prep(ed3W, 64, 32, 64, 32);

    // one GAT layer: WMMA transform -> attn coefs -> 2 atomic edge passes -> finalize
    auto gat = [&](const _Float16* Ain, int lda, int Kpad, const _Float16* Bt,
                   const float* asrc, const float* adst, const float* bias,
                   int H, int C, int sigmoid, float* dst) {
        int HC = H * C;
        dim3 g1(cdiv(N_NODES, 16), cdiv(HC, 16));
        gemm_bt_kernel<<<g1, 32, 0, stream>>>(Ain, N_NODES, lda, Kpad, Bt, HC,
                                              nullptr, t_h, nullptr, HC);
        long long NH = (long long)N_NODES * H;
        att_coef_kernel<<<cdiv(NH, 256), 256, 0, stream>>>(t_h, H, C, asrc, adst, as_b, ad_b);
        fill_kernel<<<cdiv(NH, 256), 256, 0, stream>>>(m_b, NH, -1e30f);
        fill_kernel<<<cdiv(NH, 256), 256, 0, stream>>>(s_b, NH, 0.f);
        long long NHC = (long long)N_NODES * HC;
        fill_kernel<<<cdiv(NHC, 256), 256, 0, stream>>>(agg, NHC, 0.f);
        long long EH = (long long)(N_EDGES + N_NODES) * H;
        gat_edge_max_kernel<<<cdiv(EH, 256), 256, 0, stream>>>(edge_src, edge_dst, H,
                                                               as_b, ad_b, elr, m_b);
        gat_edge_accum_kernel<<<cdiv(EH, 256), 256, 0, stream>>>(edge_src, edge_dst, H, C,
                                                                 elr, m_b, t_h, s_b, agg);
        gat_finalize_kernel<<<cdiv(NHC, 256), 256, 0, stream>>>(agg, s_b, bias, H, C,
                                                                sigmoid, dst);
    };
    auto bn_node = [&](float* X, int C, const float* g, const float* b,
                       float* Of, _Float16* Oh) {
        bn_stats_kernel<<<C, 256, 0, stream>>>(X, nullptr, N_NODES, C, bmean, brstd);
        long long tot = (long long)N_NODES * C;
        bn_apply_kernel<<<cdiv(tot, 256), 256, 0, stream>>>(X, nullptr, tot, C, bmean,
                                                            brstd, g, b, 1, Of, Oh);
    };
    auto bn_edge = [&](_Float16* X, int C, const float* g, const float* b) {
        bn_stats_kernel<<<C, 256, 0, stream>>>(nullptr, X, N_EDGES, C, bmean, brstd);
        long long tot = (long long)N_EDGES * C;
        bn_apply_kernel<<<cdiv(tot, 256), 256, 0, stream>>>(nullptr, X, tot, C, bmean,
                                                            brstd, g, b, 1, nullptr, X);
    };

    // ---- encode ----
    embed_concat_kernel<<<cdiv((long long)N_NODES * 32, 256), 256, 0, stream>>>(
        node_ids, node_feats, emb, x16);
    gat(x16, 32, 32, e1t, e1as, e1ad, e1b, HEADS, HID, 0, t_h);            // enc1
    bn_node(t_h, 128, bn1g, bn1b, nullptr, actA);                          // x1 f16
    gat(actA, 128, 128, e2t, e2as, e2ad, e2b, 1, HID, 0, t_h);             // enc2
    bn_node(t_h, HID, bn2g, bn2b, x2f, actB);                              // x2
    gat(actB, 32, 32, e3t, e3as, e3ad, e3b, 1, HID, 0, t_h);               // enc3
    bn_node(t_h, HID, bn3g, bn3b, x3f, nullptr);                           // x3
    residual_kernel<<<cdiv((long long)N_NODES * HID, 256), 256, 0, stream>>>(
        x3f, x2f, hres, hres16);                                           // h = x3+x2

    dim3 gz(cdiv(N_NODES, 16), 2);
    gemm_bt_kernel<<<gz, 32, 0, stream>>>(hres16, N_NODES, 32, 32, zmt, LAT,
                                          zmB, mu, nullptr, LAT);
    gemm_bt_kernel<<<gz, 32, 0, stream>>>(hres16, N_NODES, 32, 32, zvt, LAT,
                                          zvB, lvar, nullptr, LAT);
    fill_kernel<<<1, 256, 0, stream>>>(out_kl, 1, 0.f);
    reparam_kernel<<<cdiv((long long)N_NODES * LAT, 256), 256, 0, stream>>>(
        mu, lvar, eps, zf, z16, out_kl);

    // ---- node decode ----
    gat(z16, 32, 32, d1t, d1as, d1ad, d1b, HEADS, HID, 0, t_h);            // dec1
    bn_node(t_h, 128, dbn1g, dbn1b, nullptr, actA);
    gat(actA, 128, 128, d2t, d2as, d2ad, d2b, 1, HID, 0, t_h);             // dec2
    bn_node(t_h, HID, dbn2g, dbn2b, nullptr, actB);                        // d2 f16
    gat(actB, 32, 32, d3t, d3as, d3ad, d3b, 1, IN_CH - 1, 1, out_cont);    // dec3+sigmoid
    dim3 gc(cdiv(N_NODES, 16), 16);
    gemm_bt_kernel<<<gc, 32, 0, stream>>>(actB, N_NODES, 32, 32, cidt, 256,
                                          cidB, out_canid, nullptr, 256);

    // ---- edge decode (original edges, no self-loops) ----
    dim3 ge1(cdiv(N_EDGES, 16), 8);
    edge_gemm1_kernel<<<ge1, 32, 0, stream>>>(zf, edge_src, edge_dst, ed1t, ed1b, t1);
    bn_edge(t1, 128, edbn1g, edbn1b);
    dim3 ge2(cdiv(N_EDGES, 16), 4);
    gemm_bt_kernel<<<ge2, 32, 0, stream>>>(t1, N_EDGES, 128, 128, ed2t, 64,
                                           ed2b, nullptr, t2, 64);
    bn_edge(t2, 64, edbn2g, edbn2b);
    dim3 ge3(cdiv(N_EDGES, 16), 2);
    gemm_bt_kernel<<<ge3, 32, 0, stream>>>(t2, N_EDGES, 64, 64, ed3t, 32,
                                           ed3b, nullptr, t3, 32);
    bn_edge(t3, 32, edbn3g, edbn3b);
    edge_out_kernel<<<cdiv(N_EDGES, 256), 256, 0, stream>>>(t3, ed4W, ed4b, out_edge);
}
